// GraphSAGELayer_50766513439457
// MI455X (gfx1250) — matrix operand order repaired and checked by reading, not compiled, not run on hardware
//
#include <hip/hip_runtime.h>
#include <hip/hip_bf16.h>

typedef __attribute__((ext_vector_type(16))) __bf16 v16bf;
typedef __attribute__((ext_vector_type(8)))  __bf16 v8bf;
typedef __attribute__((ext_vector_type(8)))  float  v8f;

#define N_NODES 100000
#define N_EDGES 1600000
#define IN_DIM  128
#define OUT_DIM 256
#define EPS_F   1e-12f

// ---------------- workspace layout (bytes) ----------------
// agg   : [N_NODES, IN_DIM] f32   @ 0            (51,200,000 B)
// wsum  : [N_NODES] f32           @ 51,200,000   (400,000 B)  (contiguous with agg)
// h16   : [N_NODES, 2*IN_DIM] bf16@ 51,600,000   (51,200,000 B, 16B aligned)
// W16   : [OUT_DIM, 2*IN_DIM] bf16@ 102,800,000  (131,072 B, 32B aligned)
#define OFF_AGG   ((size_t)0)
#define OFF_WSUM  ((size_t)51200000)
#define OFF_H16   ((size_t)51600000)
#define OFF_W16   ((size_t)102800000)

// ---------------- 1) zero agg + wsum (contiguous) ----------------
__global__ void zero_f32(float* __restrict__ p, int n) {
    int i = blockIdx.x * blockDim.x + threadIdx.x;
    if (i < n) p[i] = 0.0f;
}

// ---------------- 2) W f32 -> bf16 ----------------
__global__ void conv_w(const float* __restrict__ W, __bf16* __restrict__ W16, int n) {
    int i = blockIdx.x * blockDim.x + threadIdx.x;
    if (i < n) W16[i] = (__bf16)W[i];
}

// ---------------- 3) edge scatter: 32 lanes per edge ----------------
// agg[src] += x[dst] * w ; wsum[src] += w   (f32 atomics, resolved in L2)
__global__ __launch_bounds__(256) void scatter_edges(
        const float* __restrict__ x,
        const int*   __restrict__ ei,      // [2, N_EDGES] row-major
        const float* __restrict__ ew,
        float* __restrict__ agg,
        float* __restrict__ wsum) {
    const int lane = threadIdx.x & 31;
    const int eloc = threadIdx.x >> 5;
    const int e    = blockIdx.x * 8 + eloc;
    if (e >= N_EDGES) return;

    const int   src = ei[e];
    const int   dst = ei[N_EDGES + e];
    const float w   = ew[e];

    const float4 xv = *(const float4*)(x + (size_t)dst * IN_DIM + lane * 4);
    float* a = agg + (size_t)src * IN_DIM + lane * 4;
    __hip_atomic_fetch_add(a + 0, xv.x * w, __ATOMIC_RELAXED, __HIP_MEMORY_SCOPE_AGENT);
    __hip_atomic_fetch_add(a + 1, xv.y * w, __ATOMIC_RELAXED, __HIP_MEMORY_SCOPE_AGENT);
    __hip_atomic_fetch_add(a + 2, xv.z * w, __ATOMIC_RELAXED, __HIP_MEMORY_SCOPE_AGENT);
    __hip_atomic_fetch_add(a + 3, xv.w * w, __ATOMIC_RELAXED, __HIP_MEMORY_SCOPE_AGENT);
    if (lane == 0)
        __hip_atomic_fetch_add(wsum + src, w, __ATOMIC_RELAXED, __HIP_MEMORY_SCOPE_AGENT);
}

// ---------------- 4) build h16 = bf16([x | agg/max(wsum,eps)]) ----------------
__global__ __launch_bounds__(256) void build_h(
        const float* __restrict__ x,
        const float* __restrict__ agg,
        const float* __restrict__ wsum,
        __bf16* __restrict__ h16) {
    const int row = blockIdx.x;      // one row per block
    const int k   = threadIdx.x;     // 0..255
    float v;
    if (k < IN_DIM) {
        v = x[(size_t)row * IN_DIM + k];
    } else {
        const float inv = 1.0f / fmaxf(wsum[row], EPS_F);
        v = agg[(size_t)row * IN_DIM + (k - IN_DIM)] * inv;
    }
    h16[(size_t)row * OUT_DIM + k] = (__bf16)v;
}

// ---------------- 5) GEMM: out = relu(h @ W^T + b), bf16 WMMA ----------------
// One wave owns a full 16-row m-block and all 16 n-tiles: A fragment is loaded
// once per k-block (no 16x h16 re-read), B sweeps all of W16 (L0/L2 resident).
// 128 WMMAs per wave; 6250 m-tiles = 625 blocks x 10 waves (exact, EXEC all-1s).
__global__ __launch_bounds__(320) void gemm_wmma(
        const __bf16* __restrict__ h16,   // [N_NODES, 256]
        const __bf16* __restrict__ W16,   // [256, 256] row-major (n,k)
        const float*  __restrict__ bias,  // [256]
        float*        __restrict__ out) { // [N_NODES, 256]
    const int lane  = threadIdx.x & 31;
    const int wave  = threadIdx.x >> 5;
    const int mtile = blockIdx.x * 10 + wave;  // 0..6249
    const int l15   = lane & 15;
    const int half  = lane >> 4;

    // A fragment base: row m = mtile*16 + l15, k-origin = half*16
    const __bf16* Ap = h16 + (size_t)(mtile * 16 + l15) * OUT_DIM + half * 16;
    // B fragment base for n-tile 0: W row n = l15, k-origin = half*16
    const __bf16* Bp = W16 + (size_t)l15 * OUT_DIM + half * 16;

    v8f c[16];
    #pragma unroll
    for (int nt = 0; nt < 16; ++nt) {
        const float bv = bias[nt * 16 + l15];
        c[nt] = (v8f){bv, bv, bv, bv, bv, bv, bv, bv};
    }

    #pragma unroll
    for (int kb = 0; kb < 8; ++kb) {           // K = 8 * 32 = 256
        union { v16bf v; v8bf h[2]; } a;
        a.h[0] = *(const v8bf*)(Ap + kb * 32);        // K = kb*32 + half*16 + [0..7]
        a.h[1] = *(const v8bf*)(Ap + kb * 32 + 8);    // K = kb*32 + half*16 + 8 + [0..7]
        #pragma unroll
        for (int nt = 0; nt < 16; ++nt) {
            // W row (nt*16 + l15), K = kb*32 + half*16 + [0..15]
            v16bf b = *(const v16bf*)(Bp + (size_t)nt * 16 * OUT_DIM + kb * 32);
            c[nt] = __builtin_amdgcn_wmma_f32_16x16x32_bf16(
                        /*neg_a=*/false, a.v, /*neg_b=*/false, b,
                        /*c_mod=*/(short)0, c[nt], /*reuse_a=*/false, /*reuse_b=*/false);
        }
    }

    // C/D layout: VGPR r -> m = half*8 + r, n = l15 (within each n-tile)
    #pragma unroll
    for (int nt = 0; nt < 16; ++nt) {
        float* op = out + (size_t)(mtile * 16 + half * 8) * OUT_DIM + nt * 16 + l15;
        #pragma unroll
        for (int r = 0; r < 8; ++r) {
            const float v = c[nt][r];
            op[(size_t)r * OUT_DIM] = v > 0.0f ? v : 0.0f;
        }
    }
}

extern "C" void kernel_launch(void* const* d_in, const int* in_sizes, int n_in,
                              void* d_out, int out_size, void* d_ws, size_t ws_size,
                              hipStream_t stream) {
    const float* x  = (const float*)d_in[0];  // [N_NODES, IN_DIM]
    const int*   ei = (const int*)  d_in[1];  // [2, N_EDGES]
    const float* ew = (const float*)d_in[2];  // [N_EDGES]
    const float* W  = (const float*)d_in[3];  // [OUT_DIM, 2*IN_DIM]
    const float* b  = (const float*)d_in[4];  // [OUT_DIM]
    float* out = (float*)d_out;

    char*   ws   = (char*)d_ws;
    float*  agg  = (float*)(ws + OFF_AGG);
    float*  wsum = (float*)(ws + OFF_WSUM);
    __bf16* h16  = (__bf16*)(ws + OFF_H16);
    __bf16* W16  = (__bf16*)(ws + OFF_W16);

    // 1) zero agg+wsum (contiguous region of N*128 + N floats)
    {
        int n = N_NODES * IN_DIM + N_NODES;
        zero_f32<<<(n + 255) / 256, 256, 0, stream>>>(agg, n);
    }
    // 2) convert W to bf16
    {
        int n = OUT_DIM * 2 * IN_DIM;
        conv_w<<<(n + 255) / 256, 256, 0, stream>>>(W, W16, n);
    }
    // 3) scatter: 8 edges per 256-thread block
    scatter_edges<<<N_EDGES / 8, 256, 0, stream>>>(x, ei, ew, agg, wsum);
    // 4) build concatenated bf16 feature matrix
    build_h<<<N_NODES, 256, 0, stream>>>(x, agg, wsum, h16);
    // 5) WMMA GEMM + bias + relu: 625 blocks x 10 waves = 6250 m-tiles
    gemm_wmma<<<625, 320, 0, stream>>>(h16, W16, b, out);
}
